// SelfAttention_72937134620849
// MI455X (gfx1250) — matrix operand order repaired and checked by reading, not compiled, not run on hardware
//
#include <hip/hip_runtime.h>

// ---------------------------------------------------------------------------
// Multi-head self-attention for MI455X (gfx1250, wave32, WMMA 16x16x32 bf16).
//
// ~155 GFLOP total, ~16MB true HBM I/O; all bf16 intermediates (~133MB) live in
// the 192MB L2. Attention is flash-style (S^T = K*Q^T, O^T = V^T*P) so softmax
// reductions are lane-local. This revision shares K/V tiles across the 8 waves
// of a block through LDS, fed by double-buffered async global->LDS DMA
// (global_load_async_to_lds_b128 + s_wait_asynccnt), raising L2 arithmetic
// intensity from ~16 to ~128 FLOP/byte. LDS rows are padded 16B to spread the
// 16-lane fragment reads across banks.
// ---------------------------------------------------------------------------

typedef __attribute__((ext_vector_type(16))) __bf16        v16bf;
typedef __attribute__((ext_vector_type(8)))  float         v8f;
typedef __attribute__((ext_vector_type(8)))  unsigned int  v8u;

#define WMMA_BF16(a, b, c) \
  __builtin_amdgcn_wmma_f32_16x16x32_bf16(false, (a), false, (b), (short)0, (c), false, false)

constexpr int EMB   = 128;
constexpr int HEADS = 8;
constexpr int B_    = 8;
constexpr int T_    = 2048;
constexpr int BT    = B_ * T_;        // 16384 tokens
constexpr int HE    = HEADS * EMB;    // 1024

// padded LDS strides (elements): +8 bf16 = +16B keeps 16B alignment, rotates banks
constexpr int KLD = EMB + 8;          // 136
constexpr int VLD = 32 + 8;           // 40

// ---- CDNA5 async global->LDS copy (ASYNCcnt) -------------------------------
__device__ __forceinline__ unsigned lds_addr(const void* p) {
  // addrspace(3) pointers are raw LDS byte offsets — what VDST of the async op wants
  return (unsigned)(unsigned long long)(const __attribute__((address_space(3))) void*)p;
}
__device__ __forceinline__ void async_copy_b128(const void* lds_dst, const void* gsrc) {
  asm volatile("global_load_async_to_lds_b128 %0, %1, off"
               :: "v"(lds_addr(lds_dst)), "v"((unsigned long long)gsrc)
               : "memory");
}
__device__ __forceinline__ void wait_async0() {
  asm volatile("s_wait_asynccnt 0x0" ::: "memory");
}

// ---- fragment loaders (contiguous k order k = 16*hi + s, same for A and B) ------------------
// A-matrix 16x32 bf16: lane m=l&15 holds row m; VGPR v holds K pair at
// kb = hi*16 + 2v (hi = lane>>4).
__device__ __forceinline__ v16bf ld_fragA(const __bf16* tile, int stride, int lane) {
  const int m = lane & 15, hi = (lane >> 4) & 1;
  const __bf16* rp = tile + (size_t)m * stride;
  v8u u;
#pragma unroll
  for (int v = 0; v < 8; ++v) {
    const int kb = hi * 16 + v * 2;
    u[v] = *(const unsigned*)(rp + kb);
  }
  return __builtin_bit_cast(v16bf, u);
}

// B-matrix 32x16 bf16: lane n=l&15 holds column n; lanes 0-15 hold K=0-15,
// lanes 16-31 hold K=16-31; VGPR v -> K pair at kb = hi*16 + 2v. "tile" points
// at row n0 of the row-major *transposed* operand.
__device__ __forceinline__ v16bf ld_fragB(const __bf16* tile, int stride, int lane) {
  const int n = lane & 15, hi = (lane >> 4) & 1;
  const __bf16* rp = tile + (size_t)n * stride;
  v8u u;
#pragma unroll
  for (int v = 0; v < 8; ++v) {
    const int kb = hi * 16 + v * 2;
    u[v] = *(const unsigned*)(rp + kb);
  }
  return __builtin_bit_cast(v16bf, u);
}

__device__ __forceinline__ unsigned pack_bf16(float a, float b) {
  unsigned short ua = __builtin_bit_cast(unsigned short, (__bf16)a);
  unsigned short ub = __builtin_bit_cast(unsigned short, (__bf16)b);
  return (unsigned)ua | ((unsigned)ub << 16);
}

// ---- precision conversion / weight transpose -------------------------------
__global__ void k_cvt_f32_bf16(const float* __restrict__ src, __bf16* __restrict__ dst, int n) {
  int i = blockIdx.x * blockDim.x + threadIdx.x;
  if (i < n) dst[i] = (__bf16)src[i];
}

__global__ void k_transpose_bf16(const float* __restrict__ src, __bf16* __restrict__ dst,
                                 int R, int C) {
  int i = blockIdx.x * blockDim.x + threadIdx.x;
  if (i < R * C) {
    int r = i / C, c = i % C;
    dst[(size_t)c * R + r] = (__bf16)src[i];
  }
}

// ---- QKV projection: [BT,128] x [128,1024] (x3), outputs head-major bf16 ---
// Q,K stored [b,h,t,128]; V stored transposed [b,h,128,t] for the O^T=V^T*P step.
__global__ void __launch_bounds__(128) k_qkv(
    const __bf16* __restrict__ xbf,
    const __bf16* __restrict__ Wtq, const __bf16* __restrict__ Wtk,
    const __bf16* __restrict__ Wtv,
    __bf16* __restrict__ Qbf, __bf16* __restrict__ Kbf, __bf16* __restrict__ Vt) {
  const int lane = threadIdx.x & 31;
  const int hi   = (lane >> 4) & 1;
  const int gw   = blockIdx.x * (blockDim.x >> 5) + (threadIdx.x >> 5);
  const int NT = HE / 16;            // 64 col tiles
  const int MT = BT / 16;            // 1024 row tiles
  const int mat = gw / (MT * NT);
  const int rem = gw % (MT * NT);
  const int mt = rem / NT, nt = rem % NT;
  const __bf16* Wt = (mat == 0) ? Wtq : (mat == 1) ? Wtk : Wtv;

  v8f acc;
#pragma unroll
  for (int r = 0; r < 8; ++r) acc[r] = 0.f;
#pragma unroll
  for (int ec = 0; ec < EMB / 32; ++ec) {
    v16bf a = ld_fragA(xbf + (size_t)(mt * 16) * EMB + ec * 32, EMB, lane);
    v16bf b = ld_fragB(Wt  + (size_t)(nt * 16) * EMB + ec * 32, EMB, lane);
    acc = WMMA_BF16(a, b, acc);
  }

  const int col = nt * 16 + (lane & 15);
  const int h = col >> 7, e = col & 127;
#pragma unroll
  for (int r = 0; r < 8; ++r) {
    const int row = mt * 16 + r + 8 * hi;      // token index in [0, BT)
    const int bb = row / T_, t = row % T_;
    const __bf16 val = (__bf16)acc[r];
    if (mat == 2)
      Vt[((size_t)(bb * HEADS + h) * EMB + e) * T_ + t] = val;
    else if (mat == 0)
      Qbf[((size_t)(bb * HEADS + h) * T_ + t) * EMB + e] = val;
    else
      Kbf[((size_t)(bb * HEADS + h) * T_ + t) * EMB + e] = val;
  }
}

// ---- flash attention -------------------------------------------------------
// Block = 256 threads = 8 waves, all on one (b,h); wave w owns q-tile g*8+w.
// K/V 32-key chunks staged in double-buffered LDS via async DMA.
__global__ void __launch_bounds__(256) k_attn(
    const __bf16* __restrict__ Qbf, const __bf16* __restrict__ Kbf,
    const __bf16* __restrict__ Vt, __bf16* __restrict__ Obf) {
  __shared__ __align__(16) __bf16 sK[2][32 * KLD];    // 17.0 KB
  __shared__ __align__(16) __bf16 sV[2][128 * VLD];   // 20.0 KB

  const int lane = threadIdx.x & 31;
  const int hi   = (lane >> 4) & 1;
  const int w    = threadIdx.x >> 5;
  const int bh   = blockIdx.x >> 4;               // 16 blocks per (b,h)
  const int qt   = (blockIdx.x & 15) * 8 + w;     // this wave's query tile
  const int bb   = bh >> 3, h = bh & 7;

  const __bf16* Kbase = Kbf + (size_t)bh * T_ * EMB;
  const __bf16* Vbase = Vt  + (size_t)bh * EMB * T_;

  // cooperative async stage of one 32-key chunk (K: 32x128, V^T: 128x32)
  auto issue_chunk = [&](int kc, int buf) {
    {  // K tile: 256 x 16B segments, one per thread
      const int s = threadIdx.x, row = s >> 3, seg = s & 7;
      async_copy_b128(&sK[buf][row * KLD + seg * 8],
                      Kbase + ((size_t)(kc * 32 + row)) * EMB + seg * 8);
    }
#pragma unroll
    for (int t = 0; t < 2; ++t) {  // V tile: 512 x 16B segments, two per thread
      const int s = threadIdx.x + t * 256, row = s >> 2, seg = s & 3;
      async_copy_b128(&sV[buf][row * VLD + seg * 8],
                      Vbase + (size_t)row * T_ + kc * 32 + seg * 8);
    }
  };

  // Q rows as B-operand (lane n holds query row qt*16+n), preloaded once.
  const __bf16* Qrow = Qbf + ((size_t)bh * T_ + qt * 16) * EMB;
  v16bf qf[4];
#pragma unroll
  for (int ec = 0; ec < 4; ++ec) qf[ec] = ld_fragB(Qrow + ec * 32, EMB, lane);

  v8f o[8];                                       // O^T accumulators
#pragma unroll
  for (int i = 0; i < 8; ++i)
#pragma unroll
    for (int r = 0; r < 8; ++r) o[i][r] = 0.f;

  float m_run = -1e30f, l_run = 0.f;
  const float SCL = 0.088388347648318447f * 1.4426950408889634f;  // 1/sqrt(128)*log2(e)

  constexpr int NCH = T_ / 32;                    // 64 chunks
  issue_chunk(0, 0);

  for (int kc = 0; kc < NCH; ++kc) {
    const int cur = kc & 1;
    wait_async0();         // own DMA into buf[cur] complete
    __syncthreads();       // everyone's DMA done; everyone done reading buf[cur^1]
    if (kc + 1 < NCH) issue_chunk(kc + 1, cur ^ 1);  // prefetch overlaps compute

    // S^T = K * Q^T : two 16x16 tiles (k 0..15 / 16..31), 8 WMMAs from LDS
    v8f s0, s1;
#pragma unroll
    for (int r = 0; r < 8; ++r) { s0[r] = 0.f; s1[r] = 0.f; }
#pragma unroll
    for (int ec = 0; ec < 4; ++ec) {
      v16bf a0 = ld_fragA(&sK[cur][ec * 32], KLD, lane);
      v16bf a1 = ld_fragA(&sK[cur][16 * KLD + ec * 32], KLD, lane);
      s0 = WMMA_BF16(a0, qf[ec], s0);
      s1 = WMMA_BF16(a1, qf[ec], s1);
    }

    // online softmax: lane owns query q=l&15; halves combine via shfl_xor(16)
    float cmax = -1e30f;
#pragma unroll
    for (int r = 0; r < 8; ++r) {
      s0[r] *= SCL; s1[r] *= SCL;
      cmax = fmaxf(cmax, fmaxf(s0[r], s1[r]));
    }
    cmax = fmaxf(cmax, __shfl_xor(cmax, 16, 32));
    const float m_new = fmaxf(m_run, cmax);
    const float alpha = exp2f(m_run - m_new);
    float psum = 0.f;
#pragma unroll
    for (int r = 0; r < 8; ++r) {
      s0[r] = exp2f(s0[r] - m_new);
      s1[r] = exp2f(s1[r] - m_new);
      psum += s0[r] + s1[r];
    }
    psum += __shfl_xor(psum, 16, 32);
    l_run = l_run * alpha + psum;
    m_run = m_new;
#pragma unroll
    for (int i = 0; i < 8; ++i)
#pragma unroll
      for (int r = 0; r < 8; ++r) o[i][r] *= alpha;

    // Repack P (C-layout P^T tiles) into B-operand [32k x 16q]: B wants lane
    // (q,hi) to hold K = hi*16+2v+eh; swap quarters across halves (shfl_xor 16).
    unsigned u0[4], u1[4];
#pragma unroll
    for (int v = 0; v < 4; ++v) {
      u0[v] = pack_bf16(s0[2 * v], s0[2 * v + 1]);
      u1[v] = pack_bf16(s1[2 * v], s1[2 * v + 1]);
    }
    v8u pu;
#pragma unroll
    for (int v = 0; v < 4; ++v) {
      const unsigned x0 = __shfl_xor(u0[v], 16, 32);
      const unsigned x1 = __shfl_xor(u1[v], 16, 32);
      pu[v]     = hi ? x1    : u0[v];
      pu[v + 4] = hi ? u1[v] : x0;
    }
    const v16bf pf = __builtin_bit_cast(v16bf, pu);

    // O^T += V^T * P : 8 e-tiles, 8 WMMAs from LDS
#pragma unroll
    for (int et = 0; et < 8; ++et) {
      v16bf a = ld_fragA(&sV[cur][(et * 16) * VLD], VLD, lane);
      o[et] = WMMA_BF16(a, pf, o[et]);
    }
  }

  // normalize and store as [b, t, h*128] bf16 for the output GEMM
  const float inv = 1.0f / l_run;
  const int q = qt * 16 + (lane & 15);
  __bf16* dst = Obf + ((size_t)bb * T_ + q) * HE + h * EMB;
#pragma unroll
  for (int et = 0; et < 8; ++et) {
#pragma unroll
    for (int r = 0; r < 8; r += 2) {
      const int e = et * 16 + 8 * hi + r;
      *(unsigned*)(dst + e) = pack_bf16(o[et][r] * inv, o[et][r + 1] * inv);
    }
  }
}

// ---- output projection: [BT,1024] x [1024,128] + bias, f32 out -------------
__global__ void __launch_bounds__(128) k_proj(
    const __bf16* __restrict__ Obf, const __bf16* __restrict__ Wot,
    const float* __restrict__ bo, float* __restrict__ out) {
  const int lane = threadIdx.x & 31;
  const int hi   = (lane >> 4) & 1;
  const int gw   = blockIdx.x * (blockDim.x >> 5) + (threadIdx.x >> 5);
  const int NT = EMB / 16;                       // 8 col tiles
  const int mt = gw / NT, nt = gw % NT;

  v8f acc;
#pragma unroll
  for (int r = 0; r < 8; ++r) acc[r] = 0.f;
#pragma unroll 4
  for (int ec = 0; ec < HE / 32; ++ec) {         // 32 K-steps
    v16bf a = ld_fragA(Obf + (size_t)(mt * 16) * HE + ec * 32, HE, lane);
    v16bf b = ld_fragB(Wot + (size_t)(nt * 16) * HE + ec * 32, HE, lane);
    acc = WMMA_BF16(a, b, acc);
  }
  const int col = nt * 16 + (lane & 15);
  const float bias = bo[col];
#pragma unroll
  for (int r = 0; r < 8; ++r) {
    const int row = mt * 16 + r + 8 * hi;
    out[(size_t)row * EMB + col] = acc[r] + bias;
  }
}

// ---------------------------------------------------------------------------
extern "C" void kernel_launch(void* const* d_in, const int* in_sizes, int n_in,
                              void* d_out, int out_size, void* d_ws, size_t ws_size,
                              hipStream_t stream) {
  const float* x  = (const float*)d_in[0];
  const float* Wq = (const float*)d_in[1];
  const float* Wk = (const float*)d_in[2];
  const float* Wv = (const float*)d_in[3];
  const float* Wo = (const float*)d_in[4];
  const float* bo = (const float*)d_in[5];
  float* out = (float*)d_out;

  char* ws = (char*)d_ws;
  size_t off = 0;
  auto wsalloc = [&](size_t bytes) -> void* {
    void* p = ws + off;
    off += (bytes + 255) & ~(size_t)255;
    return p;
  };
  __bf16* xbf = (__bf16*)wsalloc((size_t)BT * EMB * 2);       // 4 MiB
  __bf16* Wtq = (__bf16*)wsalloc((size_t)HE * EMB * 2);       // 256 KiB each
  __bf16* Wtk = (__bf16*)wsalloc((size_t)HE * EMB * 2);
  __bf16* Wtv = (__bf16*)wsalloc((size_t)HE * EMB * 2);
  __bf16* Wot = (__bf16*)wsalloc((size_t)HE * EMB * 2);
  __bf16* Qbf = (__bf16*)wsalloc((size_t)BT * HE * 2);        // 32 MiB each
  __bf16* Kbf = (__bf16*)wsalloc((size_t)BT * HE * 2);
  __bf16* Vtb = (__bf16*)wsalloc((size_t)BT * HE * 2);
  __bf16* Obf = (__bf16*)wsalloc((size_t)BT * HE * 2);

  // precision conversion + weight transposes (bf16)
  k_cvt_f32_bf16<<<(BT * EMB + 255) / 256, 256, 0, stream>>>(x, xbf, BT * EMB);
  k_transpose_bf16<<<(EMB * HE + 255) / 256, 256, 0, stream>>>(Wq, Wtq, EMB, HE);
  k_transpose_bf16<<<(EMB * HE + 255) / 256, 256, 0, stream>>>(Wk, Wtk, EMB, HE);
  k_transpose_bf16<<<(EMB * HE + 255) / 256, 256, 0, stream>>>(Wv, Wtv, EMB, HE);
  k_transpose_bf16<<<(HE * EMB + 255) / 256, 256, 0, stream>>>(Wo, Wot, HE, EMB);

  // QKV: 3 * (1024 row tiles * 64 col tiles) waves, 4 waves per block
  k_qkv<<<3 * (BT / 16) * (HE / 16) / 4, 128, 0, stream>>>(xbf, Wtq, Wtk, Wtv,
                                                           Qbf, Kbf, Vtb);
  // attention: 64 (b,h) * 16 blocks, 8 waves/block, one q-tile per wave
  k_attn<<<B_ * HEADS * 16, 256, 0, stream>>>(Qbf, Kbf, Vtb, Obf);
  // output projection: 1024 * 8 = 8192 waves
  k_proj<<<(BT / 16) * (EMB / 16) / 4, 128, 0, stream>>>(Obf, Wot, bo, out);
}
